// DilatedAttention_65180423685294
// MI455X (gfx1250) — matrix operand order repaired and checked by reading, not compiled, not run on hardware
//
#include <hip/hip_runtime.h>

// ---------------- problem constants (fixed shapes from the reference) -------
#define WSZ   7
#define NH    8
#define HD    32
#define BATCH 8
#define IMG   112
#define CCH   256
#define NTOK  (IMG * IMG)        // 12544
#define GH    (IMG / WSZ)        // 16 windows per axis
#define NWIN  (BATCH * GH * GH)  // 2048 windows
#define NWT   (WSZ * WSZ)        // 49 tokens per window

// LDS strides (in halfs) padded to break 64-bank conflicts
#define QSTR 264   // row-major Q/K: 64 rows x 264
#define VSTR 72    // V transposed: 256 channels x 72 rows
#define PSTR 72    // per-wave P scratch: 16 rows x 72

typedef __attribute__((ext_vector_type(16))) _Float16 v16h;
typedef __attribute__((ext_vector_type(8)))  _Float16 v8h;
typedef __attribute__((ext_vector_type(8)))  float    v8f;
typedef __attribute__((ext_vector_type(4)))  float    f32x4;

// Build a 16-half WMMA fragment from two 16-byte LDS chunks.
static __device__ inline v16h frag16(const _Float16* p0, const _Float16* p1) {
  v8h lo = *(const v8h*)p0;
  v8h hi = *(const v8h*)p1;
  return __builtin_shufflevector(lo, hi, 0, 1, 2, 3, 4, 5, 6, 7,
                                         8, 9, 10, 11, 12, 13, 14, 15);
}

// ------------------ windowed attention + fused LePE -------------------------
__global__ __launch_bounds__(256)
void win_attn_kernel(const float* __restrict__ qkv,
                     const float* __restrict__ wgt,
                     const float* __restrict__ bias,
                     float* __restrict__ out) {
  __shared__ __align__(32) _Float16 sQ[64 * QSTR];      // 33792 B, pre-scaled
  __shared__ __align__(32) _Float16 sK[64 * QSTR];      // 33792 B
  __shared__ __align__(32) _Float16 sVT[CCH * VSTR];    // 36864 B, transposed
  __shared__ __align__(32) _Float16 sP[NH * 16 * PSTR]; // 18432 B, per-wave P

  const int tid = threadIdx.x;
  const int w   = blockIdx.x;       // window id
  const int b   = w >> 8;
  const int wy  = (w >> 4) & 15;
  const int wx  = w & 15;
  const size_t SB = (size_t)BATCH * NTOK * CCH;
  const float scale = 0.17677669529663687f;  // 32^-0.5
  const float* __restrict__ vglob = qkv + 2 * SB;  // V in f32 for LePE conv

  // ---- cooperative stage: 64 padded rows x 256 ch, f32 -> f16 --------------
  // Q/K are single-use streams -> non-temporal loads (keep the 192MB L2 for V,
  // which the fused depthwise conv re-reads 9x).
  for (int idx = tid; idx < 64 * 64; idx += 256) {
    const int r  = idx >> 6;          // padded token row 0..63
    const int c4 = (idx & 63) << 2;   // channel group of 4
    f32x4 qv = {0.f, 0.f, 0.f, 0.f}, kv = qv, vv = qv;
    if (r < NWT) {
      const int n = (wy * WSZ + r / WSZ) * IMG + wx * WSZ + (r % WSZ);
      const size_t off = ((size_t)b * NTOK + n) * CCH + c4;
      qv = __builtin_nontemporal_load((const f32x4*)(qkv + off));
      kv = __builtin_nontemporal_load((const f32x4*)(qkv + SB + off));
      vv = *(const f32x4*)(vglob + off);   // RT: wants to stay in L2
    }
    _Float16* q = &sQ[r * QSTR + c4];
    q[0] = (_Float16)(qv.x * scale); q[1] = (_Float16)(qv.y * scale);
    q[2] = (_Float16)(qv.z * scale); q[3] = (_Float16)(qv.w * scale);
    _Float16* k = &sK[r * QSTR + c4];
    k[0] = (_Float16)kv.x; k[1] = (_Float16)kv.y;
    k[2] = (_Float16)kv.z; k[3] = (_Float16)kv.w;
    sVT[(c4 + 0) * VSTR + r] = (_Float16)vv.x;
    sVT[(c4 + 1) * VSTR + r] = (_Float16)vv.y;
    sVT[(c4 + 2) * VSTR + r] = (_Float16)vv.z;
    sVT[(c4 + 3) * VSTR + r] = (_Float16)vv.w;
  }
  __syncthreads();

  // ---- per-wave (= per-head) attention -------------------------------------
  const int lane = tid & 31;
  const int head = tid >> 5;
  const int c0   = head * HD;
  const int m    = lane & 15;   // row/col index within a 16-tile
  const int kh   = lane >> 4;   // lane-half select (K split in layouts)
  _Float16* pw = &sP[head * 16 * PSTR];

  // Each lane owns exactly two output channels for the whole block:
  // preload its LePE weights/bias into registers once.
  const int ca = c0 + m;        // channel for ct=0
  const int cb = ca + 16;       // channel for ct=1
  float wa[9], wb[9];
#pragma unroll
  for (int t = 0; t < 9; ++t) {
    wa[t] = wgt[ca * 9 + t];
    wb[t] = wgt[cb * 9 + t];
  }
  const float biasA = bias[ca];
  const float biasB = bias[cb];

  for (int mt = 0; mt < 4; ++mt) {
    // Q A-fragment (16x32 f16): per-lane two contiguous 8-half chunks
    const int qrow = mt * 16 + m;
    const v16h aq = frag16(&sQ[qrow * QSTR + c0 + 16 * kh],
                           &sQ[qrow * QSTR + c0 + 16 * kh + 8]);

    // S = Q * K^T : 4 WMMAs across key tiles
    v8f s[4];
#pragma unroll
    for (int nt = 0; nt < 4; ++nt) {
      const int krow = nt * 16 + m;  // B fragment: col = lane&15 = key token
      const v16h bk = frag16(&sK[krow * QSTR + c0 + 16 * kh],
                             &sK[krow * QSTR + c0 + 16 * kh + 8]);
      v8f z = {};
      s[nt] = __builtin_amdgcn_wmma_f32_16x16x32_f16(false, aq, false, bk,
                                                     (short)0, z, false, false);
    }

    // mask padded key columns (cols 49..63 live in tile 3, lanes m>=1)
    if (m >= 1) {
#pragma unroll
      for (int j = 0; j < 8; ++j) s[3][j] = -1e30f;
    }

    // softmax: C-layout row M = mt*16 + j + 8*kh spans 16 lanes of one half
#pragma unroll
    for (int j = 0; j < 8; ++j) {
      float v0 = s[0][j], v1 = s[1][j], v2 = s[2][j], v3 = s[3][j];
      float mx = fmaxf(fmaxf(v0, v1), fmaxf(v2, v3));
      mx = fmaxf(mx, __shfl_xor(mx, 1));
      mx = fmaxf(mx, __shfl_xor(mx, 2));
      mx = fmaxf(mx, __shfl_xor(mx, 4));
      mx = fmaxf(mx, __shfl_xor(mx, 8));
      const float e0 = __expf(v0 - mx), e1 = __expf(v1 - mx);
      const float e2 = __expf(v2 - mx), e3 = __expf(v3 - mx);
      float sm = e0 + e1 + e2 + e3;
      sm += __shfl_xor(sm, 1);
      sm += __shfl_xor(sm, 2);
      sm += __shfl_xor(sm, 4);
      sm += __shfl_xor(sm, 8);
      const float inv = 1.0f / sm;
      const int lr = j + 8 * kh;  // local P row 0..15
      pw[lr * PSTR +  0 + m] = (_Float16)(e0 * inv);
      pw[lr * PSTR + 16 + m] = (_Float16)(e1 * inv);
      pw[lr * PSTR + 32 + m] = (_Float16)(e2 * inv);
      pw[lr * PSTR + 48 + m] = (_Float16)(e3 * inv);
    }

    // P A-fragments (16x64 split into two k=32 steps) via LDS round-trip
    const v16h ap0 = frag16(&pw[m * PSTR + 16 * kh],
                            &pw[m * PSTR + 16 * kh + 8]);
    const v16h ap1 = frag16(&pw[m * PSTR + 32 + 16 * kh],
                            &pw[m * PSTR + 32 + 16 * kh + 8]);

    // O = P * V : B fragments from transposed V (contiguous in token dim)
    v8f o[2];
#pragma unroll
    for (int ct = 0; ct < 2; ++ct) {
      const int vc = c0 + ct * 16 + m;  // output channel = B column
      const v16h bv0 = frag16(&sVT[vc * VSTR + 16 * kh],
                              &sVT[vc * VSTR + 16 * kh + 8]);
      const v16h bv1 = frag16(&sVT[vc * VSTR + 32 + 16 * kh],
                              &sVT[vc * VSTR + 32 + 16 * kh + 8]);
      v8f z = {};
      o[ct] = __builtin_amdgcn_wmma_f32_16x16x32_f16(false, ap0, false, bv0,
                                                     (short)0, z, false, false);
      o[ct] = __builtin_amdgcn_wmma_f32_16x16x32_f16(false, ap1, false, bv1,
                                                     (short)0, o[ct], false, false);
    }

    // ---- fused epilogue: out = attn + LePE (depthwise 3x3 on f32 V) --------
    // V neighbor reads are L2 hits (V = 103MB < 192MB L2); out is written
    // exactly once, non-temporally (never re-read).
#pragma unroll
    for (int j = 0; j < 8; ++j) {
      const int row = mt * 16 + j + 8 * kh;
      if (row < NWT) {
        const int y = wy * WSZ + row / WSZ;
        const int x = wx * WSZ + row % WSZ;
        float accA = o[0][j] + biasA;
        float accB = o[1][j] + biasB;
#pragma unroll
        for (int dy = -1; dy <= 1; ++dy) {
          const int yy = y + dy;
          if (yy < 0 || yy >= IMG) continue;
#pragma unroll
          for (int dx = -1; dx <= 1; ++dx) {
            const int xx = x + dx;
            if (xx < 0 || xx >= IMG) continue;
            const int t = (dy + 1) * 3 + (dx + 1);
            const float* p = vglob + ((size_t)b * NTOK +
                                      (size_t)yy * IMG + xx) * CCH;
            accA += wa[t] * p[ca];
            accB += wb[t] * p[cb];
          }
        }
        float* dst = out + ((size_t)b * NTOK + (size_t)y * IMG + x) * CCH;
        __builtin_nontemporal_store(accA, dst + ca);
        __builtin_nontemporal_store(accB, dst + cb);
      }
    }
  }
}

// ------------------------------- launcher -----------------------------------
extern "C" void kernel_launch(void* const* d_in, const int* in_sizes, int n_in,
                              void* d_out, int out_size, void* d_ws, size_t ws_size,
                              hipStream_t stream) {
  (void)in_sizes; (void)n_in; (void)out_size; (void)d_ws; (void)ws_size;
  const float* qkv = (const float*)d_in[0];
  const float* lw  = (const float*)d_in[1];
  const float* lb  = (const float*)d_in[2];
  float* out = (float*)d_out;

  win_attn_kernel<<<NWIN, 256, 0, stream>>>(qkv, lw, lb, out);
}
